// Attention_48275432407473
// MI455X (gfx1250) — matrix operand order repaired and checked by reading, not compiled, not run on hardware
//
#include <hip/hip_runtime.h>
#include <hip/hip_bf16.h>
#include <math.h>

typedef _Float16 v8h  __attribute__((ext_vector_type(8)));
typedef _Float16 v16h __attribute__((ext_vector_type(16)));
typedef float    v8f  __attribute__((ext_vector_type(8)));
typedef unsigned int u32x4 __attribute__((ext_vector_type(4)));
typedef int          i32x8 __attribute__((ext_vector_type(8)));
typedef int          i32x4 __attribute__((ext_vector_type(4)));

#define HIDDEN 1024
#define VDIM   512
#define NINST  32768

#if __has_builtin(__builtin_amdgcn_tensor_load_to_lds)
#define USE_TDM 1
#else
#define USE_TDM 0
#endif

__device__ __forceinline__ float fast_tanh(float x) {
    // tanh(x) = 1 - 2/(exp(2x)+1); saturates correctly at +/-1
    float e = __expf(2.0f * x);
    return 1.0f - 2.0f * __builtin_amdgcn_rcpf(e + 1.0f);
}

// ---------------------------------------------------------------------------
// Kernel 0: convert W_V (fp32 [VDIM,HIDDEN]) to f16 in workspace
// ---------------------------------------------------------------------------
__global__ __launch_bounds__(256)
void cvt_w_kernel(const float* __restrict__ wv, _Float16* __restrict__ wf) {
    int i = blockIdx.x * 256 + threadIdx.x;
    if (i < VDIM * HIDDEN) wf[i] = (_Float16)wv[i];
}

// ---------------------------------------------------------------------------
// Kernel 1: scores[n] = sum_v w[v] * tanh( sum_d E[n,d] * W_V[v,d] )
// Block = 128 threads (4 waves). Wave owns 32 rows (two 16-row tiles).
// B (W_V f16) staged into LDS by the Tensor Data Mover (TDM) with hardware
// padding, shared by all waves. f16 WMMA 16x16x32, fp32 accumulate.
// ---------------------------------------------------------------------------
__global__ __launch_bounds__(128)
void scores_kernel(const float* __restrict__ E,
                   const _Float16* __restrict__ Wf,
                   const float* __restrict__ w,
                   float* __restrict__ scores) {
    constexpr int KCH = 64;       // K per staged chunk (128 B per row of B tile)
    constexpr int PAD = 8;        // stride 72 halves = 144 B -> conflict-free frags
    __shared__ _Float16 As[4][32][KCH + PAD];     // 18432 B
    __shared__ _Float16 Bs[128][KCH + PAD];       // 18432 B

    const int wave = threadIdx.x >> 5;
    const int lane = threadIdx.x & 31;
    const int g    = lane >> 4;        // K-half select within fragments
    const int l16  = lane & 15;        // A row / B,C col within 16-tile
    const int n0   = blockIdx.x * 128 + wave * 32;

    float s0[8], s1[8];
#pragma unroll
    for (int r = 0; r < 8; ++r) { s0[r] = 0.0f; s1[r] = 0.0f; }

    for (int vtg = 0; vtg < 4; ++vtg) {           // 4 groups of 8 V-tiles (128 cols)
        v8f acc[16];                              // [2*j+m]: V-tile j, row-tile m
#pragma unroll
        for (int q = 0; q < 16; ++q)
#pragma unroll
            for (int r = 0; r < 8; ++r) acc[q][r] = 0.0f;

        for (int kc = 0; kc < HIDDEN / KCH; ++kc) {   // 16 K-chunks
            __syncthreads();   // previous chunk's LDS reads complete before overwrite

            // ---- stage B: W_V f16 [128 cols x 64 K] -> LDS ----
#if USE_TDM
            if (wave == 0) {
                // Tensor DMA descriptor (D#), 2D tile, data_size=2B.
                // HW pads LDS dest by 16B every 128B row -> 144B row stride.
                const _Float16* wsrc =
                    Wf + (size_t)(vtg * 128) * HIDDEN + kc * KCH;
                uint64_t ga  = (uint64_t)(uintptr_t)wsrc;
                uint32_t lds = (uint32_t)(uintptr_t)&Bs[0][0];

                u32x4 g0;
                g0[0] = 1u;                                  // count=1 valid D#
                g0[1] = lds;                                 // lds_addr [63:32]
                g0[2] = (uint32_t)ga;                        // global_addr lo
                g0[3] = (uint32_t)(ga >> 32) | (2u << 30);   // ga hi | type=2

                i32x8 g1;
                g1[0] = (1 << 16)      // data_size = 2 bytes
                      | (1 << 20)      // pad_enable
                      | (4 << 22)      // pad_interval: every 128 B
                      | (3 << 25);     // pad_amount: 4 dwords = 16 B
                g1[1] = (KCH << 16);          // tensor_dim0[15:0] @ [63:48]
                g1[2] = (128 << 16);          // tensor_dim0 hi=0 | tensor_dim1 lo
                g1[3] = (KCH << 16);          // tensor_dim1 hi=0 | tile_dim0
                g1[4] = 128;                  // tile_dim1 | tile_dim2=0
                g1[5] = HIDDEN;               // tensor_dim0_stride[31:0]
                g1[6] = 0;                    // stride0 hi | stride1 lo
                g1[7] = 0;                    // stride1 hi

                i32x4 z4 = {0, 0, 0, 0};
                i32x8 z8 = {0, 0, 0, 0, 0, 0, 0, 0};
                __builtin_amdgcn_tensor_load_to_lds(g0, g1, z4, z4, z8, 0);
            }
#else
            {
                const _Float16* wsrc = Wf + (size_t)(vtg * 128) * HIDDEN + kc * KCH;
                for (int p = threadIdx.x; p < 128 * (KCH / 8); p += 128) {
                    int c  = p >> 3;
                    int kk = (p & 7) * 8;
                    *(uint4*)&Bs[c][kk] = *(const uint4*)(wsrc + (size_t)c * HIDDEN + kk);
                }
            }
#endif
            // ---- stage A: this wave's 32 rows x 64 K, fp32 -> f16 (vectorized) ----
            {
                const float* asrc = E + (size_t)n0 * HIDDEN + kc * KCH;
                for (int p = lane; p < 32 * (KCH / 8); p += 32) {   // 8 iters/lane
                    int row = p >> 3;         // 8 x v8f per row
                    int kk  = (p & 7) * 8;
                    v8f vf = *(const v8f*)(asrc + (size_t)row * HIDDEN + kk);
                    v8h vh = __builtin_convertvector(vf, v8h);
                    *(v8h*)&As[wave][row][kk] = vh;
                }
            }
#if USE_TDM
            if (wave == 0) __builtin_amdgcn_s_wait_tensorcnt(0);
#endif
            __syncthreads();

            // ---- 8 V-tiles x 2 K-steps; each B fragment feeds 2 WMMAs ----
#pragma unroll
            for (int j = 0; j < 8; ++j) {
#pragma unroll
                for (int ks = 0; ks < KCH / 32; ++ks) {
                    v16h b = *(const v16h*)&Bs[j * 16 + l16][ks * 32 + 16 * g];

                    const _Float16* ap0 = &As[wave][l16][ks * 32 + 16 * g];
                    const _Float16* ap1 = &As[wave][16 + l16][ks * 32 + 16 * g];
                    v8h lo0 = *(const v8h*)(ap0), hi0 = *(const v8h*)(ap0 + 8);
                    v8h lo1 = *(const v8h*)(ap1), hi1 = *(const v8h*)(ap1 + 8);
                    v16h a0, a1;
#pragma unroll
                    for (int e = 0; e < 8; ++e) {
                        a0[e] = lo0[e]; a0[e + 8] = hi0[e];
                        a1[e] = lo1[e]; a1[e + 8] = hi1[e];
                    }
                    acc[2 * j] = __builtin_amdgcn_wmma_f32_16x16x32_f16(
                        false, a0, false, b, (short)0, acc[2 * j], false, false);
                    acc[2 * j + 1] = __builtin_amdgcn_wmma_f32_16x16x32_f16(
                        false, a1, false, b, (short)0, acc[2 * j + 1], false, false);
                }
            }
        }

        // ---- epilogue for this V-group: tanh + dot with w ----
#pragma unroll
        for (int j = 0; j < 8; ++j) {
            const float wc = w[(vtg * 8 + j) * 16 + l16];
#pragma unroll
            for (int r = 0; r < 8; ++r) {
                s0[r] += wc * fast_tanh(acc[2 * j][r]);       // rows r + 8g
                s1[r] += wc * fast_tanh(acc[2 * j + 1][r]);   // rows 16 + r + 8g
            }
        }
    }

    // ---- reduce across the 16 lanes (columns) of each half-wave ----
#pragma unroll
    for (int r = 0; r < 8; ++r) {
        float a = s0[r], b = s1[r];
        a += __shfl_xor(a, 1, 32); b += __shfl_xor(b, 1, 32);
        a += __shfl_xor(a, 2, 32); b += __shfl_xor(b, 2, 32);
        a += __shfl_xor(a, 4, 32); b += __shfl_xor(b, 4, 32);
        a += __shfl_xor(a, 8, 32); b += __shfl_xor(b, 8, 32);
        s0[r] = a; s1[r] = b;
    }
    if (l16 == 0) {
#pragma unroll
        for (int r = 0; r < 8; ++r) {
            scores[n0 + 8 * g + r]      = s0[r];
            scores[n0 + 16 + 8 * g + r] = s1[r];
        }
    }
}

// ---------------------------------------------------------------------------
// Kernel 2: single-block softmax over 32768 scores -> d_out[1024:].
// Also zeroes d_out[0:1024].
// ---------------------------------------------------------------------------
__global__ __launch_bounds__(1024)
void softmax_kernel(const float* __restrict__ scores, float* __restrict__ out) {
    __shared__ float red[32];
    const int t = threadIdx.x;

    float m = -INFINITY;
    for (int i = t; i < NINST; i += 1024) m = fmaxf(m, scores[i]);
#pragma unroll
    for (int d = 1; d < 32; d <<= 1) m = fmaxf(m, __shfl_xor(m, d, 32));
    if ((t & 31) == 0) red[t >> 5] = m;
    __syncthreads();
    if (t < 32) {
        float v = red[t];
#pragma unroll
        for (int d = 1; d < 32; d <<= 1) v = fmaxf(v, __shfl_xor(v, d, 32));
        red[t] = v;
    }
    __syncthreads();
    m = red[0];
    __syncthreads();

    float ssum = 0.0f;
    for (int i = t; i < NINST; i += 1024) ssum += __expf(scores[i] - m);
#pragma unroll
    for (int d = 1; d < 32; d <<= 1) ssum += __shfl_xor(ssum, d, 32);
    if ((t & 31) == 0) red[t >> 5] = ssum;
    __syncthreads();
    if (t < 32) {
        float v = red[t];
#pragma unroll
        for (int d = 1; d < 32; d <<= 1) v += __shfl_xor(v, d, 32);
        red[t] = v;
    }
    __syncthreads();
    const float inv = 1.0f / red[0];

    for (int i = t; i < NINST; i += 1024)
        out[1024 + i] = __expf(scores[i] - m) * inv;

    out[t] = 0.0f;
}

// ---------------------------------------------------------------------------
// Kernel 3: attn[d] = sum_n p[n] * E[n,d], fp32, atomics into d_out[0:1024].
// ---------------------------------------------------------------------------
__global__ __launch_bounds__(256)
void wsum_kernel(const float* __restrict__ E,
                 const float* __restrict__ p,
                 float* __restrict__ out) {
    const int t  = threadIdx.x;
    const int r0 = blockIdx.x * (NINST / 256);
    float acc[4] = {0.f, 0.f, 0.f, 0.f};
    for (int r = 0; r < NINST / 256; ++r) {
        const int n = r0 + r;
        const float pv = p[n];
        const float* row = E + (size_t)n * HIDDEN;
#pragma unroll
        for (int i = 0; i < 4; ++i)
            acc[i] += pv * row[t + 256 * i];
    }
#pragma unroll
    for (int i = 0; i < 4; ++i)
        atomicAdd(&out[t + 256 * i], acc[i]);
}

// ---------------------------------------------------------------------------
extern "C" void kernel_launch(void* const* d_in, const int* in_sizes, int n_in,
                              void* d_out, int out_size, void* d_ws, size_t ws_size,
                              hipStream_t stream) {
    const float* E  = (const float*)d_in[0];   // [NINST, HIDDEN]
    const float* WV = (const float*)d_in[1];   // [VDIM, HIDDEN]
    const float* w  = (const float*)d_in[2];   // [VDIM]
    float* out = (float*)d_out;                // [1024 attn | 32768 weights]

    _Float16* Wf     = (_Float16*)d_ws;
    float*    scores = (float*)((char*)d_ws + (size_t)VDIM * HIDDEN * sizeof(_Float16));

    cvt_w_kernel<<<(VDIM * HIDDEN) / 256, 256, 0, stream>>>(WV, Wf);
    scores_kernel<<<NINST / 128, 128, 0, stream>>>(E, Wf, w, scores);
    softmax_kernel<<<1, 1024, 0, stream>>>(scores, out);
    wsum_kernel<<<256, 256, 0, stream>>>(E, out + 1024, out);
}
